// GCNNet_71700184039837
// MI455X (gfx1250) — matrix operand order repaired and checked by reading, not compiled, not run on hardware
//
#include <hip/hip_runtime.h>
#include <hip/hip_bf16.h>

typedef __attribute__((ext_vector_type(16))) __bf16 v16bf;
typedef __attribute__((ext_vector_type(8)))  __bf16 v8bf;
typedef __attribute__((ext_vector_type(8)))  float  v8f;
typedef __attribute__((ext_vector_type(4)))  unsigned int u32x4;
typedef __attribute__((ext_vector_type(8)))  int i32x8;
typedef __attribute__((ext_vector_type(4)))  int i32x4;

#define TILE_M 64     // rows per block (16 per wave)
#define TILE_N 128    // cols per block (8 x 16 tiles per wave)
#define TILE_K 32
#define A_ROW  40     // LDS row stride in bf16 elems: 64B data + 16B pad = 80B

#if defined(__has_builtin)
#  if __has_builtin(__builtin_amdgcn_tensor_load_to_lds) && \
      __has_builtin(__builtin_amdgcn_s_wait_tensorcnt)
#    define USE_TDM 1
#  endif
#endif
#ifndef USE_TDM
#  define USE_TDM 0
#endif

// ---------------------------------------------------------------------------
// Weight pre-swizzle: W (f32) -> bf16 in WMMA B-fragment-major layout:
//   out[ ((kb*(N/16) + nt)*32 + lane)*16 + j ],  lane=(n&15)+16*(kloc>>4), j=kloc&15
// TRANSB==0: W is [K,N] row-major.  TRANSB==1: W is [N,K] row-major (use W^T).
// ---------------------------------------------------------------------------
template<int TRANSB>
__global__ void k_swizzle_w(const float* __restrict__ W, int ld,
                            __bf16* __restrict__ out, int K, int N) {
    int idx = blockIdx.x * blockDim.x + threadIdx.x;
    if (idx >= K * N) return;
    int j    = idx & 15;
    int lane = (idx >> 4) & 31;
    int blk  = idx >> 9;
    int ntN  = N >> 4;
    int kb   = blk / ntN;
    int nt   = blk % ntN;
    int kloc = ((lane >> 4) << 4) + j;
    int k    = kb * TILE_K + kloc;
    int n    = nt * 16 + (lane & 15);
    float v  = TRANSB ? W[(long)n * ld + k] : W[(long)k * ld + n];
    out[idx] = (__bf16)v;
}

#if USE_TDM
// Issue one TDM 2D tile DMA: 64 rows x 32 bf16, global -> LDS, with 16B row
// padding (80B LDS stride) and OOB-zero rows past dim1. Fire-and-forget;
// caller pairs with s_wait_tensorcnt.
__device__ __forceinline__ void tdm_issue(const __bf16* g, const __bf16* l,
                                          unsigned dim1, int lda) {
    unsigned long long gaddr = (unsigned long long)(uintptr_t)g;
    unsigned ldsa = (unsigned)(uintptr_t)(void*)l;
    u32x4 g0;
    g0[0] = 1u;                                    // count=1 (valid user D#)
    g0[1] = ldsa;                                  // lds_addr
    g0[2] = (unsigned)gaddr;                       // global_addr[31:0]
    g0[3] = ((unsigned)(gaddr >> 32) & 0x01FFFFFFu) | (2u << 30); // addr[56:32], type=2
    i32x8 g1;
    g1[0] = (int)((1u << 16)    // data_size = 2 bytes
                | (1u << 20)    // pad_enable
                | (3u << 22)    // pad_interval = 16 DWORDs (64B)
                | (3u << 25));  // pad_amount   = 4 DWORDs (16B)
    g1[1] = (int)(32u << 16);                      // tensor_dim0 = 32
    g1[2] = (int)((dim1 & 0xffffu) << 16);         // tensor_dim1 lo
    g1[3] = (int)(((dim1 >> 16) & 0xffffu) | (32u << 16)); // dim1 hi | tile_dim0=32
    g1[4] = 64;                                    // tile_dim1 = 64 rows
    g1[5] = lda;                                   // tensor_dim0_stride
    g1[6] = 0;
    g1[7] = 0;
    i32x4 gz = (i32x4){0,0,0,0};
#if __clang_major__ >= 23
    i32x8 gz8 = (i32x8){0,0,0,0,0,0,0,0};
    __builtin_amdgcn_tensor_load_to_lds(g0, g1, gz, gz, gz8, 0);
#else
    __builtin_amdgcn_tensor_load_to_lds(g0, g1, gz, gz, 0);
#endif
}
#endif

// ---------------------------------------------------------------------------
// GEMM: C[M,N] = A[M,K] (bf16, row-major, stride lda) @ Bsw (bf16 frag-major)
// Double-buffered TDM A tiles overlap DMA with compute; 8 B fragments batch-
// preloaded from global per K-step so loadcnt waits are partial.
// ---------------------------------------------------------------------------
template<int ADD_BIAS>
__global__ __launch_bounds__(128)
void gemm_wmma_bf16(const __bf16* __restrict__ A, int lda,
                    const __bf16* __restrict__ Bsw,
                    const float* __restrict__ bias,
                    float* __restrict__ C, int ldc,
                    int M, int N, int K)
{
    __shared__ __bf16 As[2][TILE_M * A_ROW];   // 2 x 5120B double buffer

    const int tid  = threadIdx.x;
    const int lane = tid & 31;
    const int wave = tid >> 5;
    const int tileM = blockIdx.x * TILE_M;
    const int nt0   = blockIdx.y * (TILE_N / 16);
    const int ntN   = N >> 4;

    v8f acc[8];
    #pragma unroll
    for (int t = 0; t < 8; ++t) acc[t] = (v8f){0.f,0.f,0.f,0.f,0.f,0.f,0.f,0.f};

    const int m  = lane & 15;
    const int kh = lane >> 4;

#if USE_TDM
    const unsigned dim1 = (unsigned)(M - tileM);   // OOB rows read as zero
    if (__builtin_amdgcn_readfirstlane(wave) == 0) {
        tdm_issue(A + (size_t)tileM * lda, &As[0][0], dim1, lda);
        __builtin_amdgcn_s_wait_tensorcnt(0);
    }
    __syncthreads();
#endif

    int cur = 0;
    for (int k0 = 0; k0 < K; k0 += TILE_K) {
#if USE_TDM
        // async: DMA next tile into the other buffer while we compute
        if (__builtin_amdgcn_readfirstlane(wave) == 0 && (k0 + TILE_K) < K)
            tdm_issue(A + (size_t)tileM * lda + (k0 + TILE_K),
                      &As[cur ^ 1][0], dim1, lda);
#else
        __syncthreads();
        #pragma unroll
        for (int i = 0; i < 2; ++i) {
            int c   = i * 128 + tid;    // 256 chunks of 8 bf16
            int row = c >> 2;
            int ch  = c & 3;
            int gr  = tileM + row;
            v8bf v = {};
            if (gr < M) v = *(const v8bf*)&A[(long)gr * lda + k0 + ch * 8];
            *(v8bf*)&As[cur][row * A_ROW + ch * 8] = v;
        }
        __syncthreads();
#endif
        // ---- batch-preload 8 B fragments (16x global_load_b128) ----
        const v16bf* bptr = (const v16bf*)Bsw +
                            (((size_t)(k0 >> 5) * ntN + nt0) << 5) + lane;
        v16bf bfr[8];
        #pragma unroll
        for (int t = 0; t < 8; ++t) bfr[t] = bptr[t << 5];

        // ---- A fragment: two contiguous 16B LDS chunks per lane ----
        const __bf16* arow = &As[cur][(wave * 16 + m) * A_ROW + kh * 16];
        v8bf a0 = *(const v8bf*)arow;          // K = kh*16 + 0..7
        v8bf a1 = *(const v8bf*)(arow + 8);    // K = kh*16 + 8..15
        v16bf afrag = __builtin_shufflevector(a0, a1,
            0,1,2,3,4,5,6,7,8,9,10,11,12,13,14,15);

        #pragma unroll
        for (int t = 0; t < 8; ++t)
            acc[t] = __builtin_amdgcn_wmma_f32_16x16x32_bf16(
                false, afrag, false, bfr[t], (short)0, acc[t], false, false);

#if USE_TDM
        __builtin_amdgcn_s_wait_tensorcnt(0);  // no-op for waves 1..3
        __syncthreads();
#endif
        cur ^= 1;
    }

    // ---- write back C (+bias) ----
    const int rowBase = tileM + wave * 16 + ((lane >> 4) << 3);
    const int col0    = lane & 15;
    #pragma unroll
    for (int t = 0; t < 8; ++t) {
        int col = (nt0 + t) * 16 + col0;
        #pragma unroll
        for (int r = 0; r < 8; ++r) {
            int row = rowBase + r;
            if (row < M) {
                float v = acc[t][r];
                if (ADD_BIAS) v += bias[col];
                C[(long)row * ldc + col] = v;
            }
        }
    }
}

// ---------------- conversions / graph aggregation ----------------

__global__ void k_f32_to_bf16(const float* __restrict__ in,
                              __bf16* __restrict__ out, long n) {
    long i = (long)blockIdx.x * blockDim.x + threadIdx.x;
    if (i < n) out[i] = (__bf16)in[i];
}

__global__ void k_deg_init(float* deg, int n) {
    int i = blockIdx.x * blockDim.x + threadIdx.x;
    if (i < n) deg[i] = 1.0f;   // self loop
}

__global__ void k_deg_edges(float* deg, const int* __restrict__ dst, int e) {
    int i = blockIdx.x * blockDim.x + threadIdx.x;
    if (i < e) atomicAdd(&deg[dst[i]], 1.0f);
}

__global__ void k_dinv(const float* __restrict__ deg, float* dinv, int n) {
    int i = blockIdx.x * blockDim.x + threadIdx.x;
    if (i < n) dinv[i] = rsqrtf(deg[i]);
}

// agg[n, f] = dinv[n]^2 * h[n, f] + bias[f]   (self-loop term + bias)
__global__ void k_agg_init(const float* __restrict__ h,
                           const float* __restrict__ dinv,
                           const float* __restrict__ bias,
                           float* __restrict__ agg, int n, int F) {
    long idx = (long)blockIdx.x * blockDim.x + threadIdx.x;
    if (idx < (long)n * F) {
        int node = (int)(idx / F);
        int f    = (int)(idx % F);
        float di = dinv[node];
        agg[idx] = di * di * h[idx] + bias[f];
    }
}

// one block per edge: agg[dst, :] += dinv[src]*dinv[dst] * h[src, :]
__global__ void k_agg_edges(const float* __restrict__ h, int F,
                            const float* __restrict__ dinv,
                            const int* __restrict__ src,
                            const int* __restrict__ dst,
                            float* __restrict__ agg, int e) {
    int ed = blockIdx.x;
    if (ed >= e) return;
    int s = src[ed];
    int d = dst[ed];
    float coef = dinv[s] * dinv[d];
    const float* hs = h + (long)s * F;
    float* od = agg + (long)d * F;
    for (int f = threadIdx.x; f < F; f += blockDim.x)
        atomicAdd(&od[f], coef * hs[f]);
}

// relu + f32 -> bf16, writing a column slice of the concat buffer
__global__ void k_relu_bf16(const float* __restrict__ agg, int F,
                            __bf16* __restrict__ out, int ldout, int n) {
    long idx = (long)blockIdx.x * blockDim.x + threadIdx.x;
    if (idx < (long)n * F) {
        int node = (int)(idx / F);
        int f    = (int)(idx % F);
        float v = agg[idx];
        out[(long)node * ldout + f] = (__bf16)(v > 0.0f ? v : 0.0f);
    }
}

static inline int ceil_div(int a, int b) { return (a + b - 1) / b; }

extern "C" void kernel_launch(void* const* d_in, const int* in_sizes, int n_in,
                              void* d_out, int out_size, void* d_ws, size_t ws_size,
                              hipStream_t stream) {
    const float* x  = (const float*)d_in[0];
    const float* W1 = (const float*)d_in[1];
    const float* b1 = (const float*)d_in[2];
    const float* W2 = (const float*)d_in[3];
    const float* b2 = (const float*)d_in[4];
    const float* W3 = (const float*)d_in[5];
    const float* b3 = (const float*)d_in[6];
    const float* W4 = (const float*)d_in[7];   // [512,512] row-major [out,in]
    const float* b4 = (const float*)d_in[8];
    const int* edge = (const int*)d_in[9];     // [2, E]

    const int N = in_sizes[0] / 512;
    const int E = in_sizes[9] / 2;
    const int* src = edge;
    const int* dst = edge + E;

    // ---- workspace layout ----
    float*  h    = (float*)d_ws;                    // [N,256] gemm out (f32)
    float*  agg  = h + (size_t)N * 256;             // [N,256] aggregation acc
    float*  deg  = agg + (size_t)N * 256;           // [N]
    float*  dinv = deg + N;                         // [N]
    size_t  off  = ((size_t)(dinv + N) - (size_t)d_ws + 31) & ~(size_t)31;
    __bf16* xbf  = (__bf16*)((char*)d_ws + off);    // [N,512] x in bf16
    __bf16* xcbf = xbf + (size_t)N * 512;           // [N,512] concat (bf16)
    __bf16* W1s  = xcbf + (size_t)N * 512;          // swizzled weights
    __bf16* W2s  = W1s + 512 * 256;
    __bf16* W3s  = W2s + 256 * 128;
    __bf16* W4s  = W3s + 128 * 128;

    float* outp = (float*)d_out;
    const int T = 256;

    // ---- weight pre-swizzle to bf16 fragment-major ----
    k_swizzle_w<0><<<ceil_div(512 * 256, T), T, 0, stream>>>(W1, 256, W1s, 512, 256);
    k_swizzle_w<0><<<ceil_div(256 * 128, T), T, 0, stream>>>(W2, 128, W2s, 256, 128);
    k_swizzle_w<0><<<ceil_div(128 * 128, T), T, 0, stream>>>(W3, 128, W3s, 128, 128);
    k_swizzle_w<1><<<ceil_div(512 * 512, T), T, 0, stream>>>(W4, 512, W4s, 512, 512);

    // ---- x -> bf16 ; degree / normalization ----
    long nx = (long)N * 512;
    k_f32_to_bf16<<<(int)((nx + T - 1) / T), T, 0, stream>>>(x, xbf, nx);
    k_deg_init <<<ceil_div(N, T), T, 0, stream>>>(deg, N);
    k_deg_edges<<<ceil_div(E, T), T, 0, stream>>>(deg, dst, E);
    k_dinv     <<<ceil_div(N, T), T, 0, stream>>>(deg, dinv, N);

    const int gM = ceil_div(N, TILE_M);

    // ---- layer 1: h = x @ W1 ; aggregate ; relu -> xcbf[:, 0:256] ----
    {
        gemm_wmma_bf16<0><<<dim3(gM, 256 / TILE_N), 128, 0, stream>>>(
            xbf, 512, W1s, nullptr, h, 256, N, 256, 512);
        long tot = (long)N * 256;
        int  g   = (int)((tot + T - 1) / T);
        k_agg_init <<<g, T, 0, stream>>>(h, dinv, b1, agg, N, 256);
        k_agg_edges<<<E, 128, 0, stream>>>(h, 256, dinv, src, dst, agg, E);
        k_relu_bf16<<<g, T, 0, stream>>>(agg, 256, xcbf, 512, N);
    }

    // ---- layer 2: h = x1 @ W2 ; aggregate ; relu -> xcbf[:, 256:384] ----
    {
        gemm_wmma_bf16<0><<<dim3(gM, 128 / TILE_N), 128, 0, stream>>>(
            xcbf, 512, W2s, nullptr, h, 128, N, 128, 256);
        long tot = (long)N * 128;
        int  g   = (int)((tot + T - 1) / T);
        k_agg_init <<<g, T, 0, stream>>>(h, dinv, b2, agg, N, 128);
        k_agg_edges<<<E, 128, 0, stream>>>(h, 128, dinv, src, dst, agg, E);
        k_relu_bf16<<<g, T, 0, stream>>>(agg, 128, xcbf + 256, 512, N);
    }

    // ---- layer 3: h = x2 @ W3 ; aggregate ; relu -> xcbf[:, 384:512] ----
    {
        gemm_wmma_bf16<0><<<dim3(gM, 128 / TILE_N), 128, 0, stream>>>(
            xcbf + 256, 512, W3s, nullptr, h, 128, N, 128, 128);
        long tot = (long)N * 128;
        int  g   = (int)((tot + T - 1) / T);
        k_agg_init <<<g, T, 0, stream>>>(h, dinv, b3, agg, N, 128);
        k_agg_edges<<<E, 128, 0, stream>>>(h, 128, dinv, src, dst, agg, E);
        k_relu_bf16<<<g, T, 0, stream>>>(agg, 128, xcbf + 384, 512, N);
    }

    // ---- head: out = xc @ W4^T + b4 ----
    gemm_wmma_bf16<1><<<dim3(gM, 512 / TILE_N), 128, 0, stream>>>(
        xcbf, 512, W4s, b4, outp, 512, N, 512, 512);
}
